// MultiHeadAttentionWithRelPos_979252543788
// MI455X (gfx1250) — matrix operand order repaired and checked by reading, not compiled, not run on hardware
//
#include <hip/hip_runtime.h>

// Problem constants (fixed by the reference)
#define BB 2
#define SS 2048
#define DD 1024
#define HH 16
#define HD 64
#define MM (BB * SS)   // 4096

typedef __attribute__((ext_vector_type(16))) __bf16 bf16x16;
typedef __attribute__((ext_vector_type(8)))  float  floatx8;

struct U32x8 { uint4 lo; uint4 hi; };

__device__ __forceinline__ bf16x16 bf_pack(uint4 lo, uint4 hi) {
  U32x8 t; t.lo = lo; t.hi = hi;
  return __builtin_bit_cast(bf16x16, t);
}
// A-fragment: 16 consecutive bf16 (two adjacent 16B chunks)
__device__ __forceinline__ bf16x16 ld_afrag(const unsigned short* p) {
  return bf_pack(*(const uint4*)p, *(const uint4*)(p + 8));
}
// B-fragment: 16 consecutive bf16 (32 contiguous bytes)
__device__ __forceinline__ bf16x16 ld_contig(const unsigned short* p) {
  const uint4* q = (const uint4*)p;
  return bf_pack(q[0], q[1]);
}
__device__ __forceinline__ unsigned short f2bf(float f) {
  unsigned u = __builtin_bit_cast(unsigned, f);
  u += 0x7FFFu + ((u >> 16) & 1u);   // round-to-nearest-even
  return (unsigned short)(u >> 16);
}
__device__ __forceinline__ float half16_max(float t) {
  t = fmaxf(t, __shfl_xor(t, 1));
  t = fmaxf(t, __shfl_xor(t, 2));
  t = fmaxf(t, __shfl_xor(t, 4));
  t = fmaxf(t, __shfl_xor(t, 8));
  return t;
}
__device__ __forceinline__ float half16_sum(float t) {
  t += __shfl_xor(t, 1);
  t += __shfl_xor(t, 2);
  t += __shfl_xor(t, 4);
  t += __shfl_xor(t, 8);
  return t;
}
#define WMMA_BF16(a, b, c) \
  __builtin_amdgcn_wmma_f32_16x16x32_bf16(false, (a), false, (b), (short)0, (c), false, false)

// ---------------------------------------------------------------------------
// fp32 -> bf16 cast
// ---------------------------------------------------------------------------
__global__ void cast_f32_bf16(const float* __restrict__ x,
                              unsigned short* __restrict__ y, int n) {
  int i = blockIdx.x * blockDim.x + threadIdx.x;
  if (i < n) y[i] = f2bf(x[i]);
}

// ---------------------------------------------------------------------------
// GEMM: out[M,N] = X[M,K] @ W[N,K]^T + bias[N]
// M=4096, N=K=1024.  mode 0: store bf16 head-major [B,H,S,HD]
//                    mode 1: store fp32 row-major  [M,N]
// Block: 128 threads (4 waves).  Tile 64(M) x 128(N), K-step 32.
// Each wave owns a 64x32 strip: 8 WMMA per K-step, loads batched before WMMAs.
// ---------------------------------------------------------------------------
__global__ __launch_bounds__(128)
void gemm_bf16(const unsigned short* __restrict__ X,
               const unsigned short* __restrict__ W,
               const float* __restrict__ bias,
               void* __restrict__ out, int mode) {
  __shared__ unsigned short As[64 * 32];    // 4 KiB
  __shared__ unsigned short Bs[128 * 32];   // 8 KiB

  const int tid   = threadIdx.x;
  const int wave  = tid >> 5;
  const int lane  = tid & 31;
  const int lhalf = lane >> 4;     // 0: lanes 0-15, 1: lanes 16-31
  const int lcol  = lane & 15;
  const int bm    = blockIdx.y * 64;
  const int bn    = blockIdx.x * 128;

  const int lrow = tid >> 1;       // A cooperative-load row 0..63
  const int lcc  = (tid & 1) * 16; // A column chunk {0,16}

  floatx8 vz = {0.f, 0.f, 0.f, 0.f, 0.f, 0.f, 0.f, 0.f};
  floatx8 acc0[4] = {vz, vz, vz, vz};   // N cols 32w+0..15
  floatx8 acc1[4] = {vz, vz, vz, vz};   // N cols 32w+16..31

  for (int kk = 0; kk < DD; kk += 32) {
    // global tile loads into regs
    const uint4* gx = (const uint4*)(X + (size_t)(bm + lrow) * DD + kk + lcc);
    uint4 x0 = gx[0], x1 = gx[1];
    const uint4* gw = (const uint4*)(W + (size_t)(bn + tid) * DD + kk);
    uint4 w0 = gw[0], w1 = gw[1], w2 = gw[2], w3 = gw[3];
    if (kk + 32 < DD) {   // gfx1250 global_prefetch_b8
      __builtin_prefetch(X + (size_t)(bm + lrow) * DD + kk + 32 + lcc, 0, 1);
      __builtin_prefetch(W + (size_t)(bn + tid) * DD + kk + 32, 0, 1);
    }
    __syncthreads();
    { uint4* p = (uint4*)(As + lrow * 32 + lcc); p[0] = x0; p[1] = x1; }
    { uint4* p = (uint4*)(Bs + tid * 32); p[0] = w0; p[1] = w1; p[2] = w2; p[3] = w3; }
    __syncthreads();

    // batch all fragment loads, then issue 8 WMMA back-to-back
    bf16x16 bf0 = ld_contig(Bs + (32 * wave + lcol) * 32 + lhalf * 16);
    bf16x16 bf1 = ld_contig(Bs + (32 * wave + 16 + lcol) * 32 + lhalf * 16);
    bf16x16 af[4];
#pragma unroll
    for (int j = 0; j < 4; ++j)
      af[j] = ld_afrag(As + (16 * j + lcol) * 32 + lhalf * 16);
#pragma unroll
    for (int j = 0; j < 4; ++j) {
      acc0[j] = WMMA_BF16(af[j], bf0, acc0[j]);
      acc1[j] = WMMA_BF16(af[j], bf1, acc1[j]);
    }
  }

  // epilogue
#pragma unroll
  for (int t = 0; t < 2; ++t) {
    const int ncol = bn + 32 * wave + 16 * t + lcol;
    const float bv = bias[ncol];
#pragma unroll
    for (int j = 0; j < 4; ++j) {
#pragma unroll
      for (int r = 0; r < 8; ++r) {
        float v = (t ? acc1[j][r] : acc0[j][r]) + bv;
        int m = bm + 16 * j + r + lhalf * 8;
        if (mode == 0) {
          int b_ = m >> 11, s_ = m & (SS - 1);
          int h_ = ncol >> 6, hd_ = ncol & (HD - 1);
          ((unsigned short*)out)[(((size_t)b_ * HH + h_) * SS + s_) * HD + hd_] = f2bf(v);
        } else {
          ((float*)out)[(size_t)m * DD + ncol] = v;
        }
      }
    }
  }
}

// ---------------------------------------------------------------------------
// Fused flash attention: out[B,S,D](bf16) = softmax(QK^T*s + relbias + mask) V
// Block = (b, h, 64-row q tile), 128 threads / 4 waves.
// Per 32-key step: 4 score WMMA + 4 PV WMMA, online softmax.
// ---------------------------------------------------------------------------
__global__ __launch_bounds__(128)
void attn_fused(const unsigned short* __restrict__ q,   // [B,H,S,HD]
                const unsigned short* __restrict__ k,
                const unsigned short* __restrict__ v,
                const unsigned char* __restrict__ mask,  // [B,S] (1 = padded)
                unsigned short* __restrict__ out) {      // [B,S,D] bf16
  __shared__ unsigned short Qs[64 * 64];
  __shared__ unsigned short Ks[32 * 64];
  __shared__ unsigned short Vts[64 * 32];   // transposed: [hd][key]
  __shared__ unsigned char  Ms[SS];
  __shared__ unsigned short Ps[4][16 * 32];

  const int tid   = threadIdx.x;
  const int wave  = tid >> 5;
  const int lane  = tid & 31;
  const int lhalf = lane >> 4;
  const int lcol  = lane & 15;
  const int q0    = blockIdx.x * 64;
  const int h     = blockIdx.y;
  const int b     = blockIdx.z;
  const size_t bh = ((size_t)b * HH + h) * SS * HD;
  const unsigned short* qp = q + bh;
  const unsigned short* kp = k + bh;
  const unsigned short* vp = v + bh;

  // stage Q tile (64x64) and this batch's padding mask
#pragma unroll
  for (int i = 0; i < 4; ++i) {
    int c = tid + 128 * i;              // 512 chunks of 8 bf16
    int row = c >> 3, c8 = (c & 7) * 8;
    *(uint4*)(Qs + row * 64 + c8) =
        *(const uint4*)(qp + (size_t)(q0 + row) * HD + c8);
  }
  *(uint4*)(Ms + tid * 16) = *(const uint4*)(mask + (size_t)b * SS + tid * 16);
  __syncthreads();

  // persistent Q A-fragments (hd 0..31 and 32..63) for this wave's 16 rows
  const unsigned short* qa = Qs + (16 * wave + lcol) * 64 + lhalf * 16;
  bf16x16 aq0 = ld_afrag(qa);
  bf16x16 aq1 = ld_afrag(qa + 32);

  floatx8 vz = {0.f, 0.f, 0.f, 0.f, 0.f, 0.f, 0.f, 0.f};
  floatx8 o[4] = {vz, vz, vz, vz};
  float mrun[8], lrun[8];
#pragma unroll
  for (int r = 0; r < 8; ++r) { mrun[r] = -3.0e38f; lrun[r] = 0.0f; }

  const float scale = 0.125f;   // 1/sqrt(64)

  for (int kc = 0; kc < SS; kc += 32) {
    // gather K tile (32x64) and V tile (to be transposed) into regs
    int row0 = (tid * 2) >> 3, c80 = ((tid * 2) & 7) * 8;
    int row1 = (tid * 2 + 1) >> 3, c81 = ((tid * 2 + 1) & 7) * 8;
    uint4 kc0 = *(const uint4*)(kp + (size_t)(kc + row0) * HD + c80);
    uint4 kc1 = *(const uint4*)(kp + (size_t)(kc + row1) * HD + c81);
    int vkey = tid >> 2, vhb = (tid & 3) * 16;
    const uint4* vsrc = (const uint4*)(vp + (size_t)(kc + vkey) * HD + vhb);
    uint4 va = vsrc[0], vb = vsrc[1];
    if (kc + 32 < SS) {
      __builtin_prefetch(kp + (size_t)(kc + 32 + row0) * HD + c80, 0, 1);
      __builtin_prefetch(vp + (size_t)(kc + 32 + vkey) * HD + vhb, 0, 1);
    }
    __syncthreads();
    *(uint4*)(Ks + row0 * 64 + c80) = kc0;
    *(uint4*)(Ks + row1 * 64 + c81) = kc1;
    {
      unsigned arr[8] = {va.x, va.y, va.z, va.w, vb.x, vb.y, vb.z, vb.w};
#pragma unroll
      for (int e = 0; e < 8; ++e) {
        Vts[(vhb + 2 * e + 0) * 32 + vkey] = (unsigned short)arr[e];
        Vts[(vhb + 2 * e + 1) * 32 + vkey] = (unsigned short)(arr[e] >> 16);
      }
    }
    __syncthreads();

    // scores: S[16q x 32keys] in two 16x16 frags; batch loads, then 4 WMMA
    floatx8 s0 = vz, s1 = vz;
    {
      bf16x16 b00 = ld_contig(Ks + (0  + lcol) * 64 + 0  + lhalf * 16);
      bf16x16 b01 = ld_contig(Ks + (0  + lcol) * 64 + 32 + lhalf * 16);
      bf16x16 b10 = ld_contig(Ks + (16 + lcol) * 64 + 0  + lhalf * 16);
      bf16x16 b11 = ld_contig(Ks + (16 + lcol) * 64 + 32 + lhalf * 16);
      s0 = WMMA_BF16(aq0, b00, s0);
      s1 = WMMA_BF16(aq0, b10, s1);
      s0 = WMMA_BF16(aq1, b01, s0);
      s1 = WMMA_BF16(aq1, b11, s1);
    }

    const int kp0 = kc + lcol, kp1 = kc + 16 + lcol;
    const bool msk0 = Ms[kp0] != 0, msk1 = Ms[kp1] != 0;

#pragma unroll
    for (int r = 0; r < 8; ++r) {
      int qpos = q0 + 16 * wave + r + lhalf * 8;
      float f0 = s0[r] * scale - 0.1f * fabsf((float)(qpos - kp0));
      float f1 = s1[r] * scale - 0.1f * fabsf((float)(qpos - kp1));
      if (msk0) f0 = -1.0e30f;
      if (msk1) f1 = -1.0e30f;
      float tm = half16_max(fmaxf(f0, f1));
      float nm = fmaxf(mrun[r], tm);
      float corr = __expf(mrun[r] - nm);
      mrun[r] = nm;
      float p0 = __expf(f0 - nm);
      float p1 = __expf(f1 - nm);
      lrun[r] = lrun[r] * corr + half16_sum(p0 + p1);
      o[0][r] *= corr; o[1][r] *= corr; o[2][r] *= corr; o[3][r] *= corr;
      s0[r] = p0; s1[r] = p1;
    }

    // C-layout P -> A-layout via per-wave LDS scratch (in-wave, dscnt-ordered)
    unsigned short* ps = &Ps[wave][0];
#pragma unroll
    for (int r = 0; r < 8; ++r) {
      int m = r + lhalf * 8;
      ps[m * 32 + lcol]      = f2bf(s0[r]);
      ps[m * 32 + 16 + lcol] = f2bf(s1[r]);
    }
    bf16x16 ap = ld_afrag(ps + lcol * 32 + lhalf * 16);

    // batch V B-frag loads, then 4 WMMA back-to-back
    bf16x16 bvf[4];
#pragma unroll
    for (int j = 0; j < 4; ++j)
      bvf[j] = ld_contig(Vts + (16 * j + lcol) * 32 + lhalf * 16);
#pragma unroll
    for (int j = 0; j < 4; ++j)
      o[j] = WMMA_BF16(ap, bvf[j], o[j]);
  }

  // epilogue: normalize, store bf16 [B,S,D]
  float inv[8];
#pragma unroll
  for (int r = 0; r < 8; ++r) inv[r] = 1.0f / lrun[r];
#pragma unroll
  for (int j = 0; j < 4; ++j) {
#pragma unroll
    for (int r = 0; r < 8; ++r) {
      int s_ = q0 + 16 * wave + r + lhalf * 8;
      int col = h * HD + 16 * j + lcol;
      out[((size_t)b * SS + s_) * DD + col] = f2bf(o[j][r] * inv[r]);
    }
  }
}

// ---------------------------------------------------------------------------
extern "C" void kernel_launch(void* const* d_in, const int* in_sizes, int n_in,
                              void* d_out, int out_size, void* d_ws, size_t ws_size,
                              hipStream_t stream) {
  const float* f_query = (const float*)d_in[0];
  const float* f_key   = (const float*)d_in[1];
  const float* f_value = (const float*)d_in[2];
  const unsigned char* maskp = (const unsigned char*)d_in[3];
  const float* f_Wq = (const float*)d_in[4];
  const float* bq   = (const float*)d_in[5];
  const float* f_Wk = (const float*)d_in[6];
  const float* bk   = (const float*)d_in[7];
  const float* f_Wv = (const float*)d_in[8];
  const float* bv   = (const float*)d_in[9];
  const float* f_Wo = (const float*)d_in[10];
  const float* bo   = (const float*)d_in[11];

  char* ws = (char*)d_ws;
  const size_t ACT = (size_t)MM * DD * 2;      // 8 MiB bf16 activation
  const size_t WB  = (size_t)DD * DD * 2;      // 2 MiB bf16 weight
  unsigned short* Xq  = (unsigned short*)(ws + 0 * ACT);
  unsigned short* Xk  = (unsigned short*)(ws + 1 * ACT);
  unsigned short* Xv  = (unsigned short*)(ws + 2 * ACT);
  unsigned short* Wqb = (unsigned short*)(ws + 3 * ACT + 0 * WB);
  unsigned short* Wkb = (unsigned short*)(ws + 3 * ACT + 1 * WB);
  unsigned short* Wvb = (unsigned short*)(ws + 3 * ACT + 2 * WB);
  unsigned short* Wob = (unsigned short*)(ws + 3 * ACT + 3 * WB);
  unsigned short* qh  = (unsigned short*)(ws + 3 * ACT + 4 * WB + 0 * ACT);
  unsigned short* kh  = (unsigned short*)(ws + 3 * ACT + 4 * WB + 1 * ACT);
  unsigned short* vh  = (unsigned short*)(ws + 3 * ACT + 4 * WB + 2 * ACT);
  unsigned short* attn = Xq;   // reuse (Xq dead after q projection)

  const int NA = MM * DD, NW = DD * DD;
  cast_f32_bf16<<<(NA + 255) / 256, 256, 0, stream>>>(f_query, Xq, NA);
  cast_f32_bf16<<<(NA + 255) / 256, 256, 0, stream>>>(f_key,   Xk, NA);
  cast_f32_bf16<<<(NA + 255) / 256, 256, 0, stream>>>(f_value, Xv, NA);
  cast_f32_bf16<<<(NW + 255) / 256, 256, 0, stream>>>(f_Wq, Wqb, NW);
  cast_f32_bf16<<<(NW + 255) / 256, 256, 0, stream>>>(f_Wk, Wkb, NW);
  cast_f32_bf16<<<(NW + 255) / 256, 256, 0, stream>>>(f_Wv, Wvb, NW);
  cast_f32_bf16<<<(NW + 255) / 256, 256, 0, stream>>>(f_Wo, Wob, NW);

  dim3 ggrid(DD / 128, MM / 64);   // (8, 64)
  gemm_bf16<<<ggrid, 128, 0, stream>>>(Xq, Wqb, bq, qh, 0);
  gemm_bf16<<<ggrid, 128, 0, stream>>>(Xk, Wkb, bk, kh, 0);
  gemm_bf16<<<ggrid, 128, 0, stream>>>(Xv, Wvb, bv, vh, 0);

  attn_fused<<<dim3(SS / 64, HH, BB), 128, 0, stream>>>(qh, kh, vh, maskp, attn);

  gemm_bf16<<<ggrid, 128, 0, stream>>>(attn, Wob, bo, d_out, 1);
  (void)in_sizes; (void)n_in; (void)out_size; (void)ws_size;
}
